// GraphConvolution_1726576857871
// MI455X (gfx1250) — matrix operand order repaired and checked by reading, not compiled, not run on hardware
//
#include <hip/hip_runtime.h>

#define D 128   // D_IN == D_OUT == 128

typedef __bf16 v16bf __attribute__((ext_vector_type(16)));
typedef float  v8f   __attribute__((ext_vector_type(8)));

// ---------------------------------------------------------------------------
// support = x @ W  (fp32-accurate via bf16x3 split on v_wmma_f32_16x16x32_bf16)
// Block: 256 threads = 8 waves. Wave w computes the 16x16 tile
//   rows [blockIdx.x*16, +16) x cols [w*16, +16).
// ---------------------------------------------------------------------------
__global__ __launch_bounds__(256) void gcn_gemm_wmma(
    const float* __restrict__ x, const float* __restrict__ w,
    float* __restrict__ support, int n_nodes)
{
  const int tid   = threadIdx.x;
  const int wave  = tid >> 5;       // 0..7 -> N tile
  const int lane  = tid & 31;
  const int row0  = blockIdx.x * 16;
  const int n0    = wave * 16;
  const int lidx  = lane & 15;      // A: row-in-tile; B/D: col-in-tile
  const int hi    = lane >> 4;      // lane half
  const int akoff = hi * 8;         // A K-base for this lane half
  const int bkoff = hi * 16;        // unused (B also indexes with akoff)

  // Unconditional loads: clamp the row, zero the values afterwards.
  const int   gm_a  = row0 + lidx;
  const int   gmr   = (gm_a < n_nodes) ? gm_a : (n_nodes - 1);
  const float zmask = (gm_a < n_nodes) ? 1.0f : 0.0f;

  const float4* __restrict__ xr = (const float4*)(x + (size_t)gmr * D);

  v8f acc = {};
  #pragma unroll
  for (int kk = 0; kk < 4; ++kk) {          // K = 128 in 4 steps of 32
    const int kb    = kk * 32;
    const int abase = (kb + akoff) >> 2;    // float4 index of first run

    // A: two contiguous 8-float runs per lane -> 4x global_load_b128
    float4 q0 = xr[abase + 0];              // K = kb+akoff   .. +3
    float4 q1 = xr[abase + 1];              // K = kb+akoff+4 .. +7
    float4 q2 = xr[abase + 4];              // K = kb+akoff+16.. +19
    float4 q3 = xr[abase + 5];              // K = kb+akoff+20.. +23
    const float af[16] = {q0.x, q0.y, q0.z, q0.w, q1.x, q1.y, q1.z, q1.w,
                          q2.x, q2.y, q2.z, q2.w, q3.x, q3.y, q3.z, q3.w};

    v16bf ah, al, bh, bl;
    #pragma unroll
    for (int e = 0; e < 16; ++e) {
      float v  = af[e] * zmask;
      __bf16 h = (__bf16)v;
      ah[e] = h;
      al[e] = (__bf16)(v - (float)h);
    }
    // B: W[kb + akoff + e + (e >= 8 ? 8 : 0)][n0 + lidx] (row-major [K][N]), bf16 hi/lo split.
    #pragma unroll
    for (int e = 0; e < 16; ++e) {
      const int k = kb + akoff + e + (e >= 8 ? 8 : 0);
      float v  = w[k * D + n0 + lidx];
      __bf16 h = (__bf16)v;
      bh[e] = h;
      bl[e] = (__bf16)(v - (float)h);
    }

    // xh*Wh + xl*Wh + xh*Wl  (xl*Wl term ~2^-16 relative, dropped)
    acc = __builtin_amdgcn_wmma_f32_16x16x32_bf16(false, ah, false, bh,
                                                  (short)0, acc, false, false);
    acc = __builtin_amdgcn_wmma_f32_16x16x32_bf16(false, al, false, bh,
                                                  (short)0, acc, false, false);
    acc = __builtin_amdgcn_wmma_f32_16x16x32_bf16(false, ah, false, bl,
                                                  (short)0, acc, false, false);
  }

  // D layout: acc[r] -> row = row0 + r + hi*8, col = n0 + lidx
  #pragma unroll
  for (int r = 0; r < 8; ++r) {
    const int gm = row0 + r + hi * 8;
    if (gm < n_nodes)
      support[(size_t)gm * D + n0 + lidx] = acc[r];
  }
}

// ---------------------------------------------------------------------------
// out[n][d] = bias[d]   (output buffer is poisoned; full init required)
// ---------------------------------------------------------------------------
__global__ __launch_bounds__(256) void gcn_bias_init(
    const float* __restrict__ bias, float* __restrict__ out, int total)
{
  int i = blockIdx.x * 256 + threadIdx.x;
  if (i < total) out[i] = bias[i & (D - 1)];
}

// ---------------------------------------------------------------------------
// Edge scatter: one wave (32 lanes) per edge.
// Lane loads float4 of support[col], scales by edge_val, atomically adds
// into out[row]. out (51.2 MB) and support (51.2 MB) are L2-resident.
// ---------------------------------------------------------------------------
__global__ __launch_bounds__(256) void gcn_edge_scatter(
    const int* __restrict__ rows, const int* __restrict__ cols,
    const float* __restrict__ vals, const float* __restrict__ support,
    float* __restrict__ out, int n_edges)
{
  const int lane = threadIdx.x & 31;
  const int e    = blockIdx.x * 8 + (threadIdx.x >> 5);   // 8 edges / block
  if (e >= n_edges) return;

  const int   r = rows[e];
  const int   c = cols[e];
  const float v = vals[e];

  const float4* __restrict__ src = (const float4*)(support + (size_t)c * D);
  float4 g = src[lane];                       // 32 lanes * 4 floats = 128

  float* dst = out + (size_t)r * D + lane * 4;
  __hip_atomic_fetch_add(dst + 0, g.x * v, __ATOMIC_RELAXED, __HIP_MEMORY_SCOPE_AGENT);
  __hip_atomic_fetch_add(dst + 1, g.y * v, __ATOMIC_RELAXED, __HIP_MEMORY_SCOPE_AGENT);
  __hip_atomic_fetch_add(dst + 2, g.z * v, __ATOMIC_RELAXED, __HIP_MEMORY_SCOPE_AGENT);
  __hip_atomic_fetch_add(dst + 3, g.w * v, __ATOMIC_RELAXED, __HIP_MEMORY_SCOPE_AGENT);
}

// ---------------------------------------------------------------------------
extern "C" void kernel_launch(void* const* d_in, const int* in_sizes, int n_in,
                              void* d_out, int out_size, void* d_ws, size_t ws_size,
                              hipStream_t stream)
{
  const float* x        = (const float*)d_in[0];   // [N_NODES, 128] f32
  const int*   edge_row = (const int*)  d_in[1];   // [E] i32
  const int*   edge_col = (const int*)  d_in[2];   // [E] i32
  const float* edge_val = (const float*)d_in[3];   // [E] f32
  const float* weight   = (const float*)d_in[4];   // [128, 128] f32
  const float* bias     = (const float*)d_in[5];   // [128] f32

  float* out     = (float*)d_out;                  // [N_NODES, 128] f32
  float* support = (float*)d_ws;                   // [N_NODES, 128] f32 scratch

  const int n_nodes = in_sizes[0] / D;
  const int n_edges = in_sizes[1];

  // 1) support = x @ W   (WMMA, bf16x3)
  const int mtiles = (n_nodes + 15) / 16;
  gcn_gemm_wmma<<<mtiles, 256, 0, stream>>>(x, weight, support, n_nodes);

  // 2) out = broadcast(bias)
  const int total = n_nodes * D;
  gcn_bias_init<<<(total + 255) / 256, 256, 0, stream>>>(bias, out, total);

  // 3) out[row] += edge_val * support[col]   (atomic scatter, L2-resident)
  gcn_edge_scatter<<<(n_edges + 7) / 8, 256, 0, stream>>>(
      edge_row, edge_col, edge_val, support, out, n_edges);
}
